// Attention_26345329394026
// MI455X (gfx1250) — matrix operand order repaired and checked by reading, not compiled, not run on hardware
//
#include <hip/hip_runtime.h>

typedef __attribute__((ext_vector_type(16))) __bf16 v16bf;
typedef __attribute__((ext_vector_type(8)))  float  v8f;
typedef unsigned u32x4 __attribute__((ext_vector_type(4)));
typedef unsigned u32x8 __attribute__((ext_vector_type(8)));

constexpr int kN   = 1024;   // sequence length
constexpr int kDim = 512;    // model dim
constexpr int kH   = 8;      // heads
constexpr int kDk  = 32;     // head dim (q/k)
constexpr int kDv  = 32;     // head dim (v)
constexpr int kR   = 32;     // NRPF
constexpr int kSplit = 4;    // key-range split for occupancy
constexpr int kSeg = kN / kSplit;          // 256 keys per block
constexpr float kScale = 0.17677669529663687f; // 32^-0.5

// ---------------- bf16 helpers (native converts) ----------------
static __device__ __forceinline__ __bf16 f_bf(float f) { return (__bf16)f; }
static __device__ __forceinline__ unsigned short bf_bits(float f) {
  return __builtin_bit_cast(unsigned short, (__bf16)f);
}
static __device__ __forceinline__ __bf16 us_bf(unsigned short s) {
  return __builtin_bit_cast(__bf16, s);
}
static __device__ __forceinline__ v8f v8f_zero() {
  v8f z;
#pragma unroll
  for (int i = 0; i < 8; ++i) z[i] = 0.f;
  return z;
}

static __device__ __forceinline__ v8f wmma_bf16(v16bf a, v16bf b, v8f c) {
  // D(16x16 f32) = A(16x32 bf16) x B(32x16 bf16) + C
  return __builtin_amdgcn_wmma_f32_16x16x32_bf16(false, a, false, b, (short)0, c, false, false);
}

// ---------------- gfx1250 async paths ----------------
// per-lane async copy, ASYNCcnt
static __device__ __forceinline__ void async_copy_b128(unsigned ldsOff, const float* g) {
  asm volatile("global_load_async_to_lds_b128 %0, %1, off" :: "v"(ldsOff), "v"(g) : "memory");
}
static __device__ __forceinline__ void wait_async0() {
  asm volatile("s_wait_asynccnt 0" ::: "memory");
}
// Tensor Data Mover: one instruction moves a 2D tile (16 lines x 1024 f32,
// line stride kN*kR elems) from global into contiguous LDS. TENSORcnt.
// D# layout per CDNA5 ISA.
static __device__ __forceinline__ void tdm_load_tile(const float* gptr, unsigned ldsOff) {
  const unsigned long long ga = (unsigned long long)(uintptr_t)gptr;
  u32x4 g0;
  g0[0] = 1u;                                   // count=1, user descriptor
  g0[1] = ldsOff;                               // lds_addr (bytes)
  g0[2] = (unsigned)ga;                         // global_addr[31:0]
  g0[3] = ((unsigned)(ga >> 32) & 0x01FFFFFFu)  // global_addr[56:32]
        | (2u << 30);                           // type = 2 ("image")
  u32x8 g1;
  g1[0] = 2u << 16;                             // data_size = 4 bytes
  g1[1] = (unsigned)(kN * kR) << 16;            // tensor_dim0[15:0] (=32768, low half)
  g1[2] = ((unsigned)(kN * kR) >> 16)           // tensor_dim0[31:16]
        | ((unsigned)kN << 16);                 // tensor_dim1[15:0] (=1024)
  g1[3] = ((unsigned)kN >> 16)                  // tensor_dim1[31:16]
        | ((unsigned)(32 * kR) << 16);          // tile_dim0 = 1024 contiguous elems
  g1[4] = 16u;                                  // tile_dim1 = 16 lines (tile_dim2 = 0)
  g1[5] = (unsigned)(kN * kR);                  // tensor_dim0_stride[31:0]
  g1[6] = 0u;                                   // dim0_stride[47:32], dim1_stride[15:0]
  g1[7] = 0u;                                   // dim1_stride[47:16]
  asm volatile("tensor_load_to_lds %0, %1" :: "s"(g0), "s"(g1) : "memory");
}
static __device__ __forceinline__ void wait_tensor0() {
  __builtin_amdgcn_s_wait_tensorcnt(0);
}

// ---- fragment loaders (wave32 WMMA operands; contiguous k = 16*half + e in both A and B) ----
// A (16x32 bf16): lane L: m=L&15; half=L>>4; element e holds k = half*16+e
static __device__ __forceinline__ v16bf load_a_us(const unsigned short* base, int mStride, int lane) {
  const int m = lane & 15, kh = lane >> 4;
  v16bf a;
#pragma unroll
  for (int e = 0; e < 16; ++e) {
    const int v = e >> 1;
    const int k = kh * 16 + e; (void)v;
    a[e] = us_bf(base[m * mStride + k]);
  }
  return a;
}
// A fragment whose rows are heads (rows >= kH are zero)
static __device__ __forceinline__ v16bf load_a_heads(const unsigned short* base, size_t headStride, int lane) {
  const int m = lane & 15, kh = lane >> 4;
  v16bf a;
#pragma unroll
  for (int e = 0; e < 16; ++e) {
    const int v = e >> 1;
    const int k = kh * 16 + e; (void)v;
    a[e] = (m < kH) ? us_bf(base[(size_t)m * headStride + k]) : us_bf((unsigned short)0);
  }
  return a;
}
// B (32x16): lane L: n=L&15; lanes 0-15 hold K=0..15, lanes 16-31 hold K=16..31
static __device__ __forceinline__ v16bf load_b_f32(const float* base, int kStride, int nStride, int lane) {
  const int n = lane & 15, kh = lane >> 4;
  v16bf b;
#pragma unroll
  for (int e = 0; e < 16; ++e) {
    const int k = kh * 16 + e;
    b[e] = f_bf(base[(size_t)k * kStride + (size_t)n * nStride]);
  }
  return b;
}
// B with contiguous K (element = base[n*nStride + k]) -> per-lane 64B runs, b128-vectorizable
static __device__ __forceinline__ v16bf load_b_f32_contig(const float* base, int nStride, int lane) {
  const int n = lane & 15, kh = lane >> 4;
  const float* p = base + (size_t)n * nStride + kh * 16;
  v16bf b;
#pragma unroll
  for (int e = 0; e < 16; ++e) b[e] = f_bf(p[e]);
  return b;
}
static __device__ __forceinline__ v16bf load_b_us(const unsigned short* base, int kStride, int nStride, int lane) {
  const int n = lane & 15, kh = lane >> 4;
  v16bf b;
#pragma unroll
  for (int e = 0; e < 16; ++e) {
    const int k = kh * 16 + e;
    b[e] = us_bf(base[(size_t)k * kStride + (size_t)n * nStride]);
  }
  return b;
}

// =====================================================================
// Kernel 1: projections + factored relative path.
//   Qc[h,i,d] = bf16( (x@Wq)*scale + rel_content_bias )
//   Qr[h,i,r] = bf16( ((x@Wq)*scale + rel_pos_bias) @ Wrk_h^T )
//   Kb[h,i,d] = bf16( x@Wk ),  Vb[h,i,d] = bf16( x@Wv )
// =====================================================================
__global__ __launch_bounds__(256) void proj_kernel(
    const float* __restrict__ x,   const float* __restrict__ Wq,
    const float* __restrict__ Wk,  const float* __restrict__ Wv,
    const float* __restrict__ Wrk, const float* __restrict__ rcb,
    const float* __restrict__ rpb,
    unsigned short* __restrict__ Qc, unsigned short* __restrict__ Qr,
    unsigned short* __restrict__ Kb, unsigned short* __restrict__ Vb)
{
  __shared__ unsigned short xs[16 * kDim];          // 16 KB
  __shared__ unsigned short qtmp[kH * 16 * kDk];    //  8 KB

  const int tid  = threadIdx.x;
  const int lane = tid & 31;
  const int wv   = tid >> 5;
  const int i0   = blockIdx.x * 16;
  const int n16  = lane & 15, kh = lane >> 4;
  const int h    = wv;

  for (int idx = tid; idx < 16 * kDim; idx += 256) {
    const int r = idx / kDim, c = idx - r * kDim;
    xs[idx] = bf_bits(x[(size_t)(i0 + r) * kDim + c]);
  }
  __syncthreads();

  const float* Ws[3] = {Wq, Wk, Wv};
#pragma unroll
  for (int which = 0; which < 3; ++which) {
    const float* W = Ws[which];
    for (int nt = 0; nt < 2; ++nt) {
      const int c0 = h * 32 + nt * 16;
      v8f acc = v8f_zero();
      for (int kk = 0; kk < kDim / 32; ++kk) {
        v16bf a = load_a_us(xs + kk * 32, kDim, lane);
        v16bf b = load_b_f32(W + (size_t)(kk * 32) * (kH * kDk) + c0, kH * kDk, 1, lane);
        acc = wmma_bf16(a, b, acc);
      }
      if (which == 0) {
        const int dl = nt * 16 + n16;
        const float cb = rcb[h * kDk + dl];
        const float pb = rpb[h * kDk + dl];
#pragma unroll
        for (int g = 0; g < 8; ++g) {
          const int m = g + 8 * kh;
          const float qv = acc[g] * kScale;
          Qc[((size_t)h * kN + (i0 + m)) * kDk + dl] = bf_bits(qv + cb);
          qtmp[(wv * 16 + m) * kDk + dl] = bf_bits(qv + pb);
        }
      } else {
        unsigned short* dst = (which == 1) ? Kb : Vb;
#pragma unroll
        for (int g = 0; g < 8; ++g) {
          const int m = g + 8 * kh;
          dst[((size_t)h * kN + (i0 + m)) * kDk + nt * 16 + n16] = bf_bits(acc[g]);
        }
      }
    }
  }

  // Qr = qtmp[16x32] @ Wrk_h^T[32x32]   (Wrk_h^T[d][r] = Wrk[r][h*32+d])
  for (int nt = 0; nt < 2; ++nt) {
    v16bf a = load_a_us(qtmp + wv * 16 * kDk, kDk, lane);
    v16bf b = load_b_f32(Wrk + (size_t)(nt * 16) * (kH * kDk) + h * kDk,
                         /*kStride=*/1, /*nStride=*/kH * kDk, lane);
    v8f acc = wmma_bf16(a, b, v8f_zero());
#pragma unroll
    for (int g = 0; g < 8; ++g) {
      const int m = g + 8 * kh;
      Qr[((size_t)h * kN + (i0 + m)) * kR + nt * 16 + n16] = bf_bits(acc[g]);
    }
  }
}

// =====================================================================
// Kernel 2: fused flash attention. Double-buffered positions stream:
//   buffer 0 <- TDM tensor_load_to_lds (TENSORcnt)
//   buffer 1 <- global_load_async_to_lds_b128 (ASYNCcnt)
// Grid = (N/16) x kSplit; one block: 16 queries, all 8 heads, 256 keys.
// =====================================================================
__global__ __launch_bounds__(256) void attn_kernel(
    const float* __restrict__ positions,
    const unsigned short* __restrict__ Qc, const unsigned short* __restrict__ Qr,
    const unsigned short* __restrict__ Kb, const unsigned short* __restrict__ Vb,
    float* __restrict__ Opart, float* __restrict__ Mpart, float* __restrict__ Lpart)
{
  __shared__ float          posBuf[2][16][32][32];  // 128 KB, f32 positions tiles
  __shared__ float          logits[kH][16][32];     //  16 KB
  __shared__ unsigned short Pbuf[kH][16][32];       //   8 KB
  __shared__ float          alphaBuf[kH][16];
  __shared__ float          lBuf[kH][16];

  const int tid  = threadIdx.x;
  const int lane = tid & 31;
  const int wv   = tid >> 5;
  const int h    = wv;
  const int i0   = blockIdx.x * 16;
  const int seg  = blockIdx.y;
  const int jbase = seg * kSeg;
  const int n16  = lane & 15, kh = lane >> 4;

  // stage positions[i0..i0+16, j0..j0+32, :] into posBuf[buf]
  auto stage = [&](int j0, int buf) {
    if (buf == 0) {
      if (wv == 0)                                  // TDM: one wave issues one op
        tdm_load_tile(positions + ((size_t)i0 * kN + j0) * kR,
                      (unsigned)(uintptr_t)&posBuf[0][0][0][0]);
    } else {                                        // per-lane async engine
      float* dst = &posBuf[1][0][0][0];
#pragma unroll
      for (int u = 0; u < 16; ++u) {
        const int s  = tid + 256 * u;               // 4096 b128 transfers
        const int q  = s & 7;
        const int j  = (s >> 3) & 31;
        const int il = s >> 8;
        const float* g = positions + ((size_t)(i0 + il) * kN + (j0 + j)) * kR + q * 4;
        async_copy_b128((unsigned)(uintptr_t)(dst + (il * 32 + j) * 32 + q * 4), g);
      }
    }
  };

  // Resident A fragments (loop-invariant)
  v16bf aQc = load_a_us(Qc + ((size_t)h * kN + i0) * kDk, kDk, lane);
  const int il0 = 2 * wv, il1 = il0 + 1;
  v16bf aR0 = load_a_heads(Qr + (size_t)(i0 + il0) * kR, (size_t)kN * kR, lane);
  v16bf aR1 = load_a_heads(Qr + (size_t)(i0 + il1) * kR, (size_t)kN * kR, lane);

  v8f o0 = v8f_zero(), o1 = v8f_zero();
  float m_run = -3.0e38f, l_run = 0.f;

  stage(jbase, 0);

  for (int cc = 0; cc < kSeg / 32; ++cc) {
    const int j0  = jbase + cc * 32;
    const int buf = cc & 1;
    if (buf == 0) wait_tensor0();          // no-op for waves with TENSORcnt==0
    else          wait_async0();
    __syncthreads();                       // buffer visible to all waves
    if (cc + 1 < kSeg / 32)
      stage(j0 + 32, buf ^ 1);             // overlap next 64KB stream with compute

    // ---- Phase A: rel logits (WMMA rows = heads) ----
#pragma unroll
    for (int p = 0; p < 2; ++p) {
      const int il = il0 + p;
      const v16bf aR = p ? aR1 : aR0;
#pragma unroll
      for (int jt = 0; jt < 2; ++jt) {
        v16bf b = load_b_f32_contig(&posBuf[buf][il][jt * 16][0], kR, lane);
        v8f s = wmma_bf16(aR, b, v8f_zero());
        if (kh == 0) {
#pragma unroll
          for (int g = 0; g < 8; ++g)
            logits[g][il][jt * 16 + n16] = s[g];
        }
      }
    }
    __syncthreads();

    // ---- Phase B: content logits (own head) ----
#pragma unroll
    for (int jt = 0; jt < 2; ++jt) {
      const unsigned short* kb = Kb + ((size_t)h * kN + j0 + jt * 16) * kDk;
      v16bf b = load_b_us(kb, /*kStride=*/1, /*nStride=*/kDk, lane);
      v8f s = wmma_bf16(aQc, b, v8f_zero());
#pragma unroll
      for (int g = 0; g < 8; ++g) {
        const int m = g + 8 * kh;
        logits[h][m][jt * 16 + n16] += s[g];
      }
    }

    // ---- Phase C: online softmax ----
    {
      const int r = lane & 15;
      const int ch = kh * 16;
      float mx = -3.0e38f;
#pragma unroll
      for (int c = 0; c < 16; ++c) mx = fmaxf(mx, logits[h][r][ch + c]);
      mx = fmaxf(mx, __shfl_xor(mx, 16, 32));
      const float m_new = fmaxf(m_run, mx);
      const float alpha = __expf(m_run - m_new);
      float s = 0.f;
#pragma unroll
      for (int c = 0; c < 16; ++c) {
        const float pv = __expf(logits[h][r][ch + c] - m_new);
        s += pv;
        Pbuf[h][r][ch + c] = bf_bits(pv);
      }
      s += __shfl_xor(s, 16, 32);
      l_run = l_run * alpha + s;
      m_run = m_new;
      if (lane < 16) alphaBuf[h][r] = alpha;
    }
#pragma unroll
    for (int g = 0; g < 8; ++g) {
      const float al = alphaBuf[h][g + 8 * kh];
      o0[g] *= al;
      o1[g] *= al;
    }

    // ---- Phase D: O += P @ V ----
    {
      v16bf aP = load_a_us(&Pbuf[h][0][0], 32, lane);
      const unsigned short* vb = Vb + ((size_t)h * kN + j0) * kDv;
      v16bf b0 = load_b_us(vb,      /*kStride=*/kDv, /*nStride=*/1, lane);
      v16bf b1 = load_b_us(vb + 16, /*kStride=*/kDv, /*nStride=*/1, lane);
      o0 = wmma_bf16(aP, b0, o0);
      o1 = wmma_bf16(aP, b1, o1);
    }
    __syncthreads();
  }

  // ---- epilogue: raw partials for this key segment ----
  if (lane < 16) {
    Mpart[((size_t)seg * kH + h) * kN + i0 + lane] = m_run;
    Lpart[((size_t)seg * kH + h) * kN + i0 + lane] = l_run;
  }
#pragma unroll
  for (int g = 0; g < 8; ++g) {
    const int m = g + 8 * kh;
    float* dst = Opart + ((size_t)seg * kN + i0 + m) * (kH * kDv) + h * kDv;
    dst[n16]      = o0[g];
    dst[16 + n16] = o1[g];
  }
}

// =====================================================================
// Kernel 2b: merge the kSplit flash partials.
// =====================================================================
__global__ __launch_bounds__(256) void combine_kernel(
    const float* __restrict__ Opart, const float* __restrict__ Mpart,
    const float* __restrict__ Lpart, float* __restrict__ Obuf)
{
  const int i = blockIdx.x;            // query row
  const int c = threadIdx.x;           // column in [0,256)
  const int h = c >> 5;
  float m[kSplit], M = -3.0e38f;
#pragma unroll
  for (int s = 0; s < kSplit; ++s) {
    m[s] = Mpart[((size_t)s * kH + h) * kN + i];
    M = fmaxf(M, m[s]);
  }
  float L = 0.f, acc = 0.f;
#pragma unroll
  for (int s = 0; s < kSplit; ++s) {
    const float w = __expf(m[s] - M);
    L   += w * Lpart[((size_t)s * kH + h) * kN + i];
    acc += w * Opart[((size_t)s * kN + i) * (kH * kDv) + c];
  }
  Obuf[(size_t)i * (kH * kDv) + c] = acc / L;
}

// =====================================================================
// Kernel 3: out = Obuf[1024,256] @ Wo[256,512] + bo
// =====================================================================
__global__ __launch_bounds__(256) void outproj_kernel(
    const float* __restrict__ Obuf, const float* __restrict__ Wo,
    const float* __restrict__ bo, float* __restrict__ out)
{
  __shared__ unsigned short ob[16 * (kH * kDv)];   // 8 KB
  const int tid  = threadIdx.x;
  const int lane = tid & 31;
  const int wv   = tid >> 5;
  const int i0   = blockIdx.x * 16;
  const int n16  = lane & 15, kh = lane >> 4;

  for (int idx = tid; idx < 16 * (kH * kDv); idx += 256) {
    const int r = idx >> 8, c = idx & 255;
    ob[idx] = bf_bits(Obuf[(size_t)(i0 + r) * (kH * kDv) + c]);
  }
  __syncthreads();

  for (int t = 0; t < 4; ++t) {
    const int c0 = (wv * 4 + t) * 16;
    v8f acc = v8f_zero();
    for (int kk = 0; kk < kH * kDv; kk += 32) {
      v16bf a = load_a_us(ob + kk, kH * kDv, lane);
      v16bf b = load_b_f32(Wo + (size_t)kk * kDim + c0, kDim, 1, lane);
      acc = wmma_bf16(a, b, acc);
    }
    const float bias = bo[c0 + n16];
#pragma unroll
    for (int g = 0; g < 8; ++g) {
      const int m = g + 8 * kh;
      out[(size_t)(i0 + m) * kDim + c0 + n16] = acc[g] + bias;
    }
  }
}

// =====================================================================
extern "C" void kernel_launch(void* const* d_in, const int* in_sizes, int n_in,
                              void* d_out, int out_size, void* d_ws, size_t ws_size,
                              hipStream_t stream) {
  (void)in_sizes; (void)n_in; (void)out_size; (void)ws_size;
  const float* x   = (const float*)d_in[0];
  const float* pos = (const float*)d_in[1];
  const float* Wq  = (const float*)d_in[2];
  const float* Wk  = (const float*)d_in[3];
  const float* Wv  = (const float*)d_in[4];
  const float* Wrk = (const float*)d_in[5];
  const float* Wo  = (const float*)d_in[6];
  const float* bo  = (const float*)d_in[7];
  const float* rcb = (const float*)d_in[8];
  const float* rpb = (const float*)d_in[9];
  float* out = (float*)d_out;

  const size_t SZ = (size_t)kH * kN * kDk;          // 262144 elems per bf16 tensor
  unsigned short* Qc = (unsigned short*)d_ws;
  unsigned short* Qr = Qc + SZ;
  unsigned short* Kb = Qr + SZ;
  unsigned short* Vb = Kb + SZ;                     // 2 MB of bf16 tensors
  float* Obuf  = (float*)(Vb + SZ);                 // [N, 256]          1 MB
  float* Opart = Obuf + (size_t)kN * (kH * kDv);    // [kSplit, N, 256]  4 MB
  float* Mpart = Opart + (size_t)kSplit * kN * (kH * kDv); // [kSplit, H, N]
  float* Lpart = Mpart + (size_t)kSplit * kH * kN;         // [kSplit, H, N]

  proj_kernel<<<dim3(kN / 16), dim3(256), 0, stream>>>(x, Wq, Wk, Wv, Wrk, rcb, rpb,
                                                       Qc, Qr, Kb, Vb);
  attn_kernel<<<dim3(kN / 16, kSplit), dim3(256), 0, stream>>>(pos, Qc, Qr, Kb, Vb,
                                                               Opart, Mpart, Lpart);
  combine_kernel<<<dim3(kN), dim3(256), 0, stream>>>(Opart, Mpart, Lpart, Obuf);
  outproj_kernel<<<dim3(kN / 16), dim3(256), 0, stream>>>(Obuf, Wo, bo, out);
}
